// DinoMask2Former_19172734009688
// MI455X (gfx1250) — 8-bit matrix operand order repaired and checked by reading, compiled, not run on hardware
//
#include <hip/hip_runtime.h>
#include <hip/hip_bf16.h>


typedef __attribute__((ext_vector_type(8))) int v8i;

constexpr int Q    = 300;
constexpr int QP   = 304;
constexpr int NT   = 19;
constexpr int HW   = 102400;
constexpr int W    = 320;
constexpr int NC   = 81;
constexpr int KSPLIT = 8;
constexpr int KCHUNK = HW / KSPLIT;

constexpr size_t OUT_KEEP   = 0;
constexpr size_t OUT_SCORE  = 300;
constexpr size_t OUT_LABEL  = 600;
constexpr size_t OUT_MASKS  = 900;
constexpr size_t OUT_BOXES  = 900 + (size_t)Q * HW;

constexpr size_t OFF_PACKED = 0;
constexpr size_t OFF_SCORES = (size_t)QP * HW;
constexpr size_t OFF_LABELS = OFF_SCORES + 1280;
constexpr size_t OFF_AREA   = OFF_LABELS + 1280;
constexpr size_t OFF_SUMP   = OFF_AREA   + 1280;
constexpr size_t OFF_MINX   = OFF_SUMP   + 1280;
constexpr size_t OFF_MAXX   = OFF_MINX   + 1280;
constexpr size_t OFF_MINY   = OFF_MAXX   + 1280;
constexpr size_t OFF_MAXY   = OFF_MINY   + 1280;
constexpr size_t OFF_ORDER  = OFF_MAXY   + 1280;
constexpr size_t OFF_AREAS  = OFF_ORDER  + 1280;
constexpr size_t OFF_VALID  = OFF_AREAS  + 1280;
constexpr size_t OFF_IOU    = OFF_VALID  + 1280;

__global__ __launch_bounds__(256) void k_binarize_pack(
    const float* __restrict__ masks, float* __restrict__ out_bin,
    unsigned char* __restrict__ packed, int* __restrict__ area,
    float* __restrict__ sumprob, int* __restrict__ minx, int* __restrict__ maxx,
    int* __restrict__ miny, int* __restrict__ maxy)
{
  const int q = blockIdx.x;
  const int tid = threadIdx.x;
  unsigned char* prow = packed + (size_t)q * HW;

  if (q >= Q) {
    uint4 z = make_uint4(0u, 0u, 0u, 0u);
    uint4* p16 = (uint4*)prow;
    for (int t = tid; t < HW / 16; t += 256) p16[t] = z;
    if (tid == 0) area[q] = 0;
    return;
  }

  const float4* in4 = (const float4*)(masks + (size_t)q * HW);
  float4* ob4 = (float4*)(out_bin + (size_t)q * HW);

  int   la = 0;    float lp = 0.0f;
  int   lx0 = W, lx1 = -1, ly0 = W, ly1 = -1;

  for (int t = tid; t < HW / 4; t += 256) {
    float4 v = in4[t];
    float vv[4] = {v.x, v.y, v.z, v.w};
    float bb[4];
    unsigned int pk = 0;
    const int p = t * 4;
#pragma unroll
    for (int k = 0; k < 4; ++k) {
      const int b = vv[k] > 0.0f;
      bb[k] = b ? 1.0f : 0.0f;
      if (b) {
        la += 1;
        lp += 1.0f / (1.0f + __expf(-vv[k]));
        const int pix = p + k;
        const int x = pix % W, y = pix / W;
        lx0 = min(lx0, x); lx1 = max(lx1, x);
        ly0 = min(ly0, y); ly1 = max(ly1, y);
        pk |= 1u << (8 * k);
      }
    }
    ob4[t] = make_float4(bb[0], bb[1], bb[2], bb[3]);
    ((unsigned int*)prow)[t] = pk;
  }

  __shared__ int   sA[256];  __shared__ float sP[256];
  __shared__ int   sX0[256], sX1[256], sY0[256], sY1[256];
  sA[tid] = la; sP[tid] = lp;
  sX0[tid] = lx0; sX1[tid] = lx1; sY0[tid] = ly0; sY1[tid] = ly1;
  __syncthreads();
  for (int s = 128; s > 0; s >>= 1) {
    if (tid < s) {
      sA[tid] += sA[tid + s];           sP[tid] += sP[tid + s];
      sX0[tid] = min(sX0[tid], sX0[tid + s]); sX1[tid] = max(sX1[tid], sX1[tid + s]);
      sY0[tid] = min(sY0[tid], sY0[tid + s]); sY1[tid] = max(sY1[tid], sY1[tid + s]);
    }
    __syncthreads();
  }
  if (tid == 0) {
    area[q] = sA[0]; sumprob[q] = sP[0];
    minx[q] = sX0[0]; maxx[q] = sX1[0]; miny[q] = sY0[0]; maxy[q] = sY1[0];
  }
}

__global__ void k_scores(const float* __restrict__ logits,
                         float* __restrict__ scores, int* __restrict__ labels)
{
  const int q = blockIdx.x * blockDim.x + threadIdx.x;
  if (q >= Q) return;
  const float* row = logits + (size_t)q * NC;
  float m = -3.0e38f, b80 = -3.0e38f; int idx = 0;
  for (int c = 0; c < NC; ++c) {
    const float v = row[c];
    m = fmaxf(m, v);
    if (c < NC - 1 && v > b80) { b80 = v; idx = c; }
  }
  float s = 0.0f;
  for (int c = 0; c < NC; ++c) s += __expf(row[c] - m);
  scores[q] = __expf(b80 - m) / s;
  labels[q] = idx;
}

__global__ __launch_bounds__(320) void k_order(
    const float* __restrict__ scores, const int* __restrict__ area,
    int* __restrict__ order, float* __restrict__ area_s, int* __restrict__ valid_s)
{
  __shared__ float s_sc[Q];
  const int tid = threadIdx.x;
  if (tid < Q) s_sc[tid] = scores[tid];
  if (tid >= Q && tid < QP) order[tid] = tid;
  __syncthreads();
  if (tid < Q) {
    const float si = s_sc[tid];
    int r = 0;
    for (int j = 0; j < Q; ++j) {
      const float sj = s_sc[j];
      r += (sj > si) || (sj == si && j < tid);
    }
    order[r] = tid;
  }
  __syncthreads();
  if (tid < QP) {
    const int q = order[tid];
    const float ar = (tid < Q) ? (float)area[q] : 0.0f;
    area_s[tid]  = ar;
    valid_s[tid] = (tid < Q) && (s_sc[q] >= 0.05f) && (ar > 0.0f);
  }
}

__global__ __launch_bounds__(32 * KSPLIT) void k_iou_gemm(
    const unsigned char* __restrict__ packed, const int* __restrict__ order,
    const float* __restrict__ area_s, float* __restrict__ iou)
{
  const int tid  = threadIdx.x;
  const int lane = tid & 31;
  const int wave = tid >> 5;
  const int half = lane >> 4;
  const int l15  = lane & 15;
  const int tm = blockIdx.x, tn = blockIdx.y;

  const int mrow = order[tm * 16 + l15];
  const int nrow = order[tn * 16 + l15];
  const unsigned char* pA = packed + (size_t)mrow * HW + half * 16;
  const unsigned char* pB = packed + (size_t)nrow * HW + half * 16;

  const int kbeg = wave * KCHUNK;
  const int kend = kbeg + KCHUNK;

  v8i acc = {0, 0, 0, 0, 0, 0, 0, 0};
#pragma unroll 4
  for (int k0 = kbeg; k0 < kend; k0 += 64) {
    const int2 a0 = *(const int2*)(pA + k0);
    const int2 a1 = *(const int2*)(pA + k0 + 8);
    const int2 a2 = *(const int2*)(pA + k0 + 32);
    const int2 a3 = *(const int2*)(pA + k0 + 40);
    const int4 b0 = *(const int4*)(pB + k0);
    const int4 b1 = *(const int4*)(pB + k0 + 32);
    v8i av, bv;
    av[0] = a0.x; av[1] = a0.y; av[2] = a1.x; av[3] = a1.y;
    av[4] = a2.x; av[5] = a2.y; av[6] = a3.x; av[7] = a3.y;
    bv[0] = b0.x; bv[1] = b0.y; bv[2] = b0.z; bv[3] = b0.w;
    bv[4] = b1.x; bv[5] = b1.y; bv[6] = b1.z; bv[7] = b1.w;
    acc = __builtin_amdgcn_wmma_i32_16x16x64_iu8(
         false, av,  false, bv, acc,
         false,  false);
  }

  __shared__ int red[KSPLIT * 256];
#pragma unroll
  for (int k = 0; k < 8; ++k) red[wave * 256 + lane * 8 + k] = acc[k];
  __syncthreads();

  if (wave == 0) {
    const int   jg = tn * 16 + l15;
    const float aj = area_s[jg];
#pragma unroll
    for (int k = 0; k < 8; ++k) {
      int s = 0;
#pragma unroll
      for (int w = 0; w < KSPLIT; ++w) s += red[w * 256 + lane * 8 + k];
      const int   ig    = tm * 16 + half * 8 + k;
      const float inter = (float)s;
      const float uni   = fmaxf(area_s[ig] + aj - inter, 1.0f);
      iou[(size_t)ig * QP + jg] = inter / uni;
    }
  }
}

__global__ __launch_bounds__(320) void k_nms_finalize(
    const float* __restrict__ iou, const int* __restrict__ valid_s,
    const int* __restrict__ order, const float* __restrict__ scores,
    const int* __restrict__ labels, const int* __restrict__ area,
    const float* __restrict__ sumprob, const int* __restrict__ minx,
    const int* __restrict__ maxx, const int* __restrict__ miny,
    const int* __restrict__ maxy, float* __restrict__ out)
{
  __shared__ int keep[QP];
  __shared__ int ovl;
  __shared__ int cnt;
  const int tid = threadIdx.x;
  if (tid < QP) keep[tid] = 0;
  if (tid == 0) cnt = 0;
  __syncthreads();

  for (int i = 0; i < Q; ++i) {
    if (tid == 0) ovl = 0;
    __syncthreads();
    if (tid < QP && keep[tid] && iou[(size_t)i * QP + tid] > 0.85f) ovl = 1;
    __syncthreads();
    if (tid == 0) {
      const int k = valid_s[i] && !ovl && (cnt < 100);
      keep[i] = k;
      cnt += k;
    }
    __syncthreads();
  }

  if (tid < Q) {
    const int q  = order[tid];
    const int kq = keep[tid];
    const int ar = area[q];
    const float quality = sumprob[q] / fmaxf((float)ar, 1.0f);
    out[OUT_KEEP  + q] = (float)kq;
    out[OUT_SCORE + q] = kq ? scores[q] * quality : 0.0f;
    out[OUT_LABEL + q] = (float)labels[q];
    float* boxes = out + OUT_BOXES;
    const int bb = kq && (ar > 0);
    boxes[q * 4 + 0] = bb ? (float)minx[q]        : 0.0f;
    boxes[q * 4 + 1] = bb ? (float)miny[q]        : 0.0f;
    boxes[q * 4 + 2] = bb ? (float)(maxx[q] + 1)  : 0.0f;
    boxes[q * 4 + 3] = bb ? (float)(maxy[q] + 1)  : 0.0f;
  }
}

extern "C" void kernel_launch(void* const* d_in, const int* in_sizes, int n_in,
                              void* d_out, int out_size, void* d_ws, size_t ws_size,
                              hipStream_t stream) {
  (void)in_sizes; (void)n_in; (void)out_size; (void)ws_size;
  const float* logits = (const float*)d_in[0];
  const float* masks  = (const float*)d_in[1];
  float* out = (float*)d_out;
  char*  ws  = (char*)d_ws;

  unsigned char* packed  = (unsigned char*)(ws + OFF_PACKED);
  float* scores  = (float*)(ws + OFF_SCORES);
  int*   labels  = (int*)  (ws + OFF_LABELS);
  int*   area    = (int*)  (ws + OFF_AREA);
  float* sumprob = (float*)(ws + OFF_SUMP);
  int*   minx    = (int*)  (ws + OFF_MINX);
  int*   maxx    = (int*)  (ws + OFF_MAXX);
  int*   miny    = (int*)  (ws + OFF_MINY);
  int*   maxy    = (int*)  (ws + OFF_MAXY);
  int*   order   = (int*)  (ws + OFF_ORDER);
  float* area_s  = (float*)(ws + OFF_AREAS);
  int*   valid_s = (int*)  (ws + OFF_VALID);
  float* iou     = (float*)(ws + OFF_IOU);

  k_binarize_pack<<<QP, 256, 0, stream>>>(masks, out + OUT_MASKS, packed, area,
                                          sumprob, minx, maxx, miny, maxy);
  k_scores<<<(Q + 31) / 32, 32, 0, stream>>>(logits, scores, labels);
  k_order<<<1, 320, 0, stream>>>(scores, area, order, area_s, valid_s);
  k_iou_gemm<<<dim3(NT, NT), 32 * KSPLIT, 0, stream>>>(packed, order, area_s, iou);
  k_nms_finalize<<<1, 320, 0, stream>>>(iou, valid_s, order, scores, labels,
                                        area, sumprob, minx, maxx, miny, maxy, out);
}
